// VitDetAttention_4964982194891
// MI455X (gfx1250) — matrix operand order repaired and checked by reading, not compiled, not run on hardware
//
#include <hip/hip_runtime.h>

#define DEV static __device__ __forceinline__

typedef __attribute__((ext_vector_type(16))) __bf16 v16bf;
typedef __attribute__((ext_vector_type(8)))  float  v8f;

typedef unsigned short bf16raw;

DEV bf16raw f2bf(float f) {
  union { float f; unsigned u; } x; x.f = f;
  unsigned u = x.u;
  unsigned r = ((u >> 16) & 1u) + 0x7FFFu;   // round-to-nearest-even
  return (bf16raw)((u + r) >> 16);
}
DEV float bf2f(bf16raw h) {
  union { unsigned u; float f; } x; x.u = ((unsigned)h) << 16;
  return x.f;
}
DEV __bf16 bfbits(bf16raw h) {
  union { unsigned short u; __bf16 b; } x; x.u = h; return x.b;
}

// Problem constants (shapes fixed by setup_inputs)
constexpr int Bc  = 8;
constexpr int Hc  = 32;
constexpr int Wc  = 32;
constexpr int Cc  = 768;
constexpr int NHc = 12;
constexpr int HDc = 64;
constexpr int HWc = Hc * Wc;        // 1024
constexpr int BQc = Bc * NHc;       // 96
constexpr int Mrows = Bc * HWc;     // 8192

// ---- WMMA operand loaders (wave32 CDNA5 layouts) ----------------------------
// A: 16x32 bf16, row-major source. lane%16 = M row (baked into row_ptr),
// lane/16 = half. elem e<8 -> K = kbase+8*half+e ; e>=8 -> K = kbase+16+8*half+(e-8)
DEV v16bf load_a_g(const bf16raw* row_ptr, int kbase, int half) {
  v16bf a;
  const bf16raw* p = row_ptr + kbase + 8 * half;
#pragma unroll
  for (int e = 0; e < 8; ++e) a[e] = bfbits(p[e]);
#pragma unroll
  for (int e = 0; e < 8; ++e) a[8 + e] = bfbits(p[16 + e]);
  return a;
}
// B: 32x16 bf16 from [N][K]-contiguous source. lane%16 = N col (baked into
// col_ptr), lane/16 = halfb. elem e<8 -> K = kbase+8*halfb+e ; e>=8 -> K = kbase+16+8*halfb+(e-8)
DEV v16bf load_b_g(const bf16raw* col_ptr, int kbase, int halfb) {
  v16bf b;
  const bf16raw* p = col_ptr + kbase + 8 * halfb;
#pragma unroll
  for (int e = 0; e < 8; ++e) b[e] = bfbits(p[e]);
#pragma unroll
  for (int e = 0; e < 8; ++e) b[8 + e] = bfbits(p[16 + e]);
  return b;
}

DEV v8f wmma_bf16(v16bf a, v16bf b, v8f c) {
  return __builtin_amdgcn_wmma_f32_16x16x32_bf16(false, a, false, b, (short)0, c,
                                                 false, false);
}

// ---- gfx1250 async global->LDS copy (ASYNCcnt path) -------------------------
DEV void async_ld_b128(unsigned lds_off, const void* src) {
  asm volatile("global_load_async_to_lds_b128 %0, %1, off"
               :: "v"(lds_off), "v"(src) : "memory");
}
DEV void wait_async_le4() { asm volatile("s_wait_asynccnt 4" ::: "memory"); }
DEV void wait_async_0()   { asm volatile("s_wait_asynccnt 0" ::: "memory"); }

// stage one contiguous 2KB key tile (16 rows x 64 dims bf16) into LDS
DEV void stage_ktile(const bf16raw* src, unsigned lds_dst, int lane) {
  const char* s = (const char*)src + lane * 64;
  unsigned d = lds_dst + lane * 64;
  async_ld_b128(d,      s);
  async_ld_b128(d + 16, s + 16);
  async_ld_b128(d + 32, s + 32);
  async_ld_b128(d + 48, s + 48);
}

// ---- small prep kernels -----------------------------------------------------
__global__ void k_cvt(const float* __restrict__ src, bf16raw* __restrict__ dst, int n) {
  int i = blockIdx.x * blockDim.x + threadIdx.x;
  if (i < n) dst[i] = f2bf(src[i]);
}

// src fp32 [K][N] -> dst bf16 [N][K]
__global__ void k_transpose_cvt(const float* __restrict__ src, bf16raw* __restrict__ dst,
                                int K, int N) {
  int i = blockIdx.x * blockDim.x + threadIdx.x;
  if (i >= K * N) return;
  int n = i / K, k = i % K;
  dst[i] = f2bf(src[(size_t)k * N + n]);
}

// ---- QKV projection: X(8192x768) @ WqkvT -> q(scaled)/k/vT, bf16 ------------
// Each wave: 16 rows x 64 cols (4 accumulators, A reused 4x). Block = 4 waves.
__global__ void __launch_bounds__(128)
k_gemm_qkv(const bf16raw* __restrict__ xb, const bf16raw* __restrict__ wt,
           const float* __restrict__ bias, bf16raw* __restrict__ qb,
           bf16raw* __restrict__ kb, bf16raw* __restrict__ vtb) {
  const int lane = threadIdx.x & 31;
  const int wid  = threadIdx.x >> 5;
  const int half = lane >> 4;
  const int nloc = lane & 15;
  const int mtile = blockIdx.y;                 // 0..511
  const int ng = blockIdx.x * 4 + wid;          // 0..35 (64-col groups)

  const bf16raw* arow  = xb + (size_t)(mtile * 16 + nloc) * Cc;
  const bf16raw* bcol0 = wt + (size_t)(ng * 64 + nloc) * Cc;

  v8f acc[4] = {v8f{}, v8f{}, v8f{}, v8f{}};
  for (int kk = 0; kk < Cc; kk += 32) {
    __builtin_prefetch(bcol0 + kk + 64, 0, 1);
    v16bf a = load_a_g(arow, kk, half);
#pragma unroll
    for (int t = 0; t < 4; ++t) {
      v16bf b = load_b_g(bcol0 + (size_t)t * 16 * Cc, kk, half);
      acc[t] = wmma_bf16(a, b, acc[t]);
    }
  }

  // this wave's 64 cols are exactly one (which, head) pair
  const int col0  = ng * 64;                    // 0..2240, step 64
  const int which = col0 / Cc;                  // 0=q 1=k 2=v (wave-uniform)
  const int head  = (col0 % Cc) / HDc;          // wave-uniform
#pragma unroll
  for (int t = 0; t < 4; ++t) {
    const int col = col0 + t * 16 + nloc;
    const int d = (col % Cc) % HDc;             // = t*16 + nloc
    const float bv = bias[col];
#pragma unroll
    for (int r = 0; r < 8; ++r) {
      int m = mtile * 16 + half * 8 + r;        // global row 0..8191
      int bidx = m >> 10;
      int hw = m & 1023;
      int bq = bidx * NHc + head;
      float v = acc[t][r] + bv;
      if (which == 0)
        qb[((size_t)bq * HWc + hw) * HDc + d] = f2bf(v * 0.125f);  // hd^-0.5
      else if (which == 1)
        kb[((size_t)bq * HWc + hw) * HDc + d] = f2bf(v);
      else
        vtb[((size_t)bq * HDc + d) * HWc + hw] = f2bf(v);          // transposed
    }
  }
}

// ---- decomposed rel-pos terms ----------------------------------------------
// relh[bq][h][w][kh], relw[bq][h][w][kw]; q is pre-scaled so multiply by 8.
__global__ void k_rel(const bf16raw* __restrict__ qb, const float* __restrict__ rph,
                      const float* __restrict__ rpw, float* __restrict__ relh,
                      float* __restrict__ relw) {
  int id = blockIdx.x * blockDim.x + threadIdx.x;
  if (id >= BQc * Hc * Wc * 32) return;
  int j = id & 31;
  int w = (id >> 5) & 31;
  int h = (id >> 10) & 31;
  int bq = id >> 15;
  const bf16raw* q = qb + ((size_t)bq * HWc + h * Wc + w) * HDc;
  const float* ph = rph + (size_t)(h - j + Hc - 1) * HDc;
  const float* pw = rpw + (size_t)(w - j + Wc - 1) * HDc;
  float ah = 0.f, aw = 0.f;
#pragma unroll 4
  for (int cidx = 0; cidx < HDc; ++cidx) {
    float qv = bf2f(q[cidx]);
    ah += qv * ph[cidx];
    aw += qv * pw[cidx];
  }
  relh[id] = ah * 8.0f;
  relw[id] = aw * 8.0f;
}

// ---- fused attention: scores + rel bias + softmax + P@V --------------------
// grid = (64 query blocks, 96 heads), 128 threads (4 waves).
// LDS: S fp32 [16][1024] | P bf16 [16][1024] | reduce | 4x double-buffer k-stage
constexpr int SMEM_S   = 0;
constexpr int SMEM_P   = 16 * 1024 * 4;                 // 65536
constexpr int SMEM_RED = SMEM_P + 16 * 1024 * 2;        // 98304
constexpr int SMEM_KST = SMEM_RED + (128 + 32) * 4;     // 98944 (128B aligned)
constexpr int SMEM_SZ  = SMEM_KST + 4 * 2 * 2048;       // 115328

__global__ void __launch_bounds__(128)
k_attn(const bf16raw* __restrict__ qb, const bf16raw* __restrict__ kb,
       const bf16raw* __restrict__ vtb, const float* __restrict__ relh,
       const float* __restrict__ relw, bf16raw* __restrict__ ob) {
  extern __shared__ unsigned char smem[];
  float* S = (float*)(smem + SMEM_S);
  bf16raw* P = (bf16raw*)(smem + SMEM_P);
  float* red = (float*)(smem + SMEM_RED);
  float* rowmax = red + 128;
  float* rowsum = rowmax + 16;

  const int lane = threadIdx.x & 31;
  const int wid  = threadIdx.x >> 5;
  const int half = lane >> 4;
  const int nloc = lane & 15;
  const int qblk = blockIdx.x;     // 0..63
  const int bq   = blockIdx.y;     // 0..95
  const int qbase = qblk * 16;
  const int h = qbase >> 5;        // 16 queries share one H row

  // per-wave double-buffered key-tile stage (2KB per buffer)
  bf16raw* stage = (bf16raw*)(smem + SMEM_KST) + wid * 2048;
  const unsigned stage_lds = (unsigned)(uintptr_t)stage;

  // A operand (q tile, 16x64) loaded once
  const bf16raw* qrow = qb + ((size_t)bq * HWc + qbase + nloc) * HDc;
  v16bf aq0 = load_a_g(qrow, 0, half);
  v16bf aq1 = load_a_g(qrow, 32, half);

  const float* relh_base = relh + (((size_t)bq * Hc + h) * Wc) * 32;
  const float* relw_base = relw + (((size_t)bq * Hc + h) * Wc) * 32;
  const bf16raw* khead = kb + (size_t)bq * HWc * HDc;

  // ---- scores: each wave covers 16 of the 64 key tiles, async double-buffer
  int buf = 0;
  stage_ktile(khead + (size_t)(wid * 16) * HDc, stage_lds, lane);
  for (int kt = wid; kt < 64; kt += 4) {
    if (kt + 4 < 64) {
      stage_ktile(khead + (size_t)((kt + 4) * 16) * HDc,
                  stage_lds + (unsigned)((buf ^ 1) * 4096), lane);
      wait_async_le4();            // current buffer's 4 b128s have landed
    } else {
      wait_async_0();
    }
    const bf16raw* kt_lds = stage + buf * 2048 + nloc * 64;  // row nloc of tile
    v16bf b0 = load_b_g(kt_lds, 0, half);
    v16bf b1 = load_b_g(kt_lds, 32, half);
    v8f c = {};
    c = wmma_bf16(aq0, b0, c);
    c = wmma_bf16(aq1, b1, c);
    const int kbase = kt * 16;
    const int kidx = kbase + nloc;
    const int kh = kidx >> 5, kw = kidx & 31;
#pragma unroll
    for (int r = 0; r < 8; ++r) {
      int ml = half * 8 + r;                 // local query row
      int wq = (qbase + ml) & 31;
      float v = c[r] + relh_base[wq * 32 + kh] + relw_base[wq * 32 + kw];
      S[ml * 1024 + kidx] = v;
    }
    buf ^= 1;
  }
  __syncthreads();

  // ---- softmax over 1024 cols; 8 threads per row
  const int row = threadIdx.x >> 3;   // 0..15
  const int seg = threadIdx.x & 7;    // 0..7
  {
    float pm = -3.0e38f;
    const float* sr = S + row * 1024 + seg * 128;
    for (int i = 0; i < 128; ++i) pm = fmaxf(pm, sr[i]);
    red[row * 8 + seg] = pm;
  }
  __syncthreads();
  if (seg == 0) {
    float m = red[row * 8];
    for (int j = 1; j < 8; ++j) m = fmaxf(m, red[row * 8 + j]);
    rowmax[row] = m;
  }
  __syncthreads();
  {
    const float rm = rowmax[row];
    float ps = 0.f;
    const float* sr = S + row * 1024 + seg * 128;
    bf16raw* pr = P + row * 1024 + seg * 128;
    for (int i = 0; i < 128; ++i) {
      float e = __expf(sr[i] - rm);
      pr[i] = f2bf(e);                  // unnormalized probs, bf16 for WMMA
      ps += e;
    }
    red[row * 8 + seg] = ps;
  }
  __syncthreads();
  if (seg == 0) {
    float s = 0.f;
    for (int j = 0; j < 8; ++j) s += red[row * 8 + j];
    rowsum[row] = s;
  }
  __syncthreads();

  // ---- P @ V : wave wid owns hd columns [wid*16, wid*16+16)
  const int ncol = wid * 16 + nloc;
  const bf16raw* vrow = vtb + ((size_t)bq * HDc + ncol) * HWc;  // contiguous K
  const bf16raw* prow = P + nloc * 1024;                        // A lane row
  v8f acc = {};
  for (int kc = 0; kc < 1024; kc += 32) {
    v16bf a = load_a_g(prow, kc, half);
    v16bf b = load_b_g(vrow, kc, half);
    acc = wmma_bf16(a, b, acc);
  }
  const int bb = bq / NHc, head = bq % NHc;
#pragma unroll
  for (int r = 0; r < 8; ++r) {
    int ml = half * 8 + r;
    float v = acc[r] / rowsum[ml];
    size_t orow = (size_t)bb * HWc + qbase + ml;
    ob[orow * Cc + head * HDc + wid * 16 + nloc] = f2bf(v);
  }
}

// ---- output projection: O(8192x768) @ WprojT + b -> fp32 -------------------
// Each wave: 16 rows x 64 cols, 4 accumulators.
__global__ void __launch_bounds__(128)
k_gemm_proj(const bf16raw* __restrict__ ob, const bf16raw* __restrict__ wt,
            const float* __restrict__ bias, float* __restrict__ out) {
  const int lane = threadIdx.x & 31;
  const int wid  = threadIdx.x >> 5;
  const int half = lane >> 4;
  const int nloc = lane & 15;
  const int mtile = blockIdx.y;                 // 0..511
  const int ng = blockIdx.x * 4 + wid;          // 0..11 (64-col groups)

  const bf16raw* arow  = ob + (size_t)(mtile * 16 + nloc) * Cc;
  const bf16raw* bcol0 = wt + (size_t)(ng * 64 + nloc) * Cc;

  v8f acc[4] = {v8f{}, v8f{}, v8f{}, v8f{}};
  for (int kk = 0; kk < Cc; kk += 32) {
    __builtin_prefetch(bcol0 + kk + 64, 0, 1);
    v16bf a = load_a_g(arow, kk, half);
#pragma unroll
    for (int t = 0; t < 4; ++t) {
      v16bf b = load_b_g(bcol0 + (size_t)t * 16 * Cc, kk, half);
      acc[t] = wmma_bf16(a, b, acc[t]);
    }
  }
#pragma unroll
  for (int t = 0; t < 4; ++t) {
    const int col = ng * 64 + t * 16 + nloc;
    const float bv = bias[col];
#pragma unroll
    for (int r = 0; r < 8; ++r) {
      int m = mtile * 16 + half * 8 + r;
      out[(size_t)m * Cc + col] = acc[t][r] + bv;
    }
  }
}

// ---- launcher ---------------------------------------------------------------
extern "C" void kernel_launch(void* const* d_in, const int* in_sizes, int n_in,
                              void* d_out, int out_size, void* d_ws, size_t ws_size,
                              hipStream_t stream) {
  (void)in_sizes; (void)n_in; (void)out_size; (void)ws_size;
  const float* hs    = (const float*)d_in[0];
  const float* qkvw  = (const float*)d_in[1];
  const float* qkvb  = (const float*)d_in[2];
  const float* projw = (const float*)d_in[3];
  const float* projb = (const float*)d_in[4];
  const float* rph   = (const float*)d_in[5];
  const float* rpw   = (const float*)d_in[6];
  float* out = (float*)d_out;

  unsigned char* ws = (unsigned char*)d_ws;
  size_t off = 0;
  auto take = [&](size_t bytes) {
    unsigned char* p = ws + off;
    off = (off + bytes + 255) & ~(size_t)255;
    return p;
  };
  bf16raw* xb     = (bf16raw*)take((size_t)Mrows * Cc * 2);        // 12.6 MB
  bf16raw* wqkvT  = (bf16raw*)take((size_t)3 * Cc * Cc * 2);       //  3.5 MB
  bf16raw* wprojT = (bf16raw*)take((size_t)Cc * Cc * 2);           //  1.2 MB
  bf16raw* qb     = (bf16raw*)take((size_t)BQc * HWc * HDc * 2);   // 12.6 MB
  bf16raw* kbuf   = (bf16raw*)take((size_t)BQc * HWc * HDc * 2);   // 12.6 MB
  bf16raw* vtb    = (bf16raw*)take((size_t)BQc * HDc * HWc * 2);   // 12.6 MB
  float*   relh   = (float*)take((size_t)BQc * Hc * Wc * 32 * 4);  // 12.6 MB
  float*   relw   = (float*)take((size_t)BQc * Hc * Wc * 32 * 4);  // 12.6 MB
  bf16raw* obuf   = (bf16raw*)take((size_t)Mrows * Cc * 2);        // 12.6 MB

  const int nX = Mrows * Cc;                         // 6291456
  k_cvt<<<(nX + 255) / 256, 256, 0, stream>>>(hs, xb, nX);
  k_transpose_cvt<<<(3 * Cc * Cc + 255) / 256, 256, 0, stream>>>(qkvw, wqkvT, Cc, 3 * Cc);
  k_transpose_cvt<<<(Cc * Cc + 255) / 256, 256, 0, stream>>>(projw, wprojT, Cc, Cc);

  // 36 64-col groups / 4 waves = 9 blocks in x
  k_gemm_qkv<<<dim3(9, 512), 128, 0, stream>>>(xb, wqkvT, qkvb, qb, kbuf, vtb);

  const int nRel = BQc * Hc * Wc * 32;               // 3145728
  k_rel<<<(nRel + 255) / 256, 256, 0, stream>>>(qb, rph, rpw, relh, relw);

  k_attn<<<dim3(64, 96), 128, SMEM_SZ, stream>>>(qb, kbuf, vtb, relh, relw, obuf);

  // 12 64-col groups / 4 waves = 3 blocks in x
  k_gemm_proj<<<dim3(3, 512), 128, 0, stream>>>(obuf, wprojT, projb, out);
}
